// RotaryEmbeddingWithNTKFlashAttention_28887950033374
// MI455X (gfx1250) — compile-verified, matrix operand order checked by reading, not run on hardware
//
#include <hip/hip_runtime.h>
#include <hip/hip_bf16.h>

// ---- problem constants -------------------------------------------------
static constexpr int Bz = 2;          // batch
static constexpr int Nn = 2048;       // sequence
static constexpr int Cc = 1024;       // channels
static constexpr int Hh = 16;         // heads
static constexpr int Dd = 64;         // head dim
static constexpr int Mrows = Bz * Nn; // 4096 GEMM rows
static constexpr float SCALE = 0.125f; // 1/sqrt(64)

typedef __bf16 bf16_t;
typedef __attribute__((ext_vector_type(8)))  __bf16 v8bf;
typedef __attribute__((ext_vector_type(16))) __bf16 v16bf;
typedef __attribute__((ext_vector_type(8)))  float  v8f;

__device__ __forceinline__ v8f v8f_zero() {
  v8f z = {0.f,0.f,0.f,0.f,0.f,0.f,0.f,0.f};
  return z;
}

// Load a 16x32 bf16 operand fragment for one lane.
// Per-lane slot->K map: slots 0..7 -> K = 8*half + i ; slots 8..15 -> K = 16 + 8*half + i
// (matches the CDNA5 16-bit A-matrix VGPR layout; used identically for A and B
//  operands so the contraction pairing is consistent.)
__device__ __forceinline__ v16bf load_frag(const bf16_t* p, int hx) {
  union { v16bf v; v8bf h[2]; } u;
  u.h[0] = *(const v8bf*)(p + 8 * hx);
  u.h[1] = *(const v8bf*)(p + 16 + 8 * hx);
  return u.v;
}

__device__ __forceinline__ v8f wmma_bf16(v16bf a, v16bf b, v8f c) {
  return __builtin_amdgcn_wmma_f32_16x16x32_bf16(
      /*neg_a=*/false, a, /*neg_b=*/false, b,
      /*c_mod=*/(short)0, c, /*reuse_a=*/false, /*reuse_b=*/false);
}

// ---- conversion kernels ------------------------------------------------
__global__ void cvt_f32_bf16(const float* __restrict__ in, bf16_t* __restrict__ out, int n) {
  int i = blockIdx.x * blockDim.x + threadIdx.x;
  if (i < n) out[i] = (bf16_t)in[i];
}

// out[c][r] = in[r][c] ; in is rows x cols f32, out is cols x rows bf16
__global__ void transpose_cvt(const float* __restrict__ in, bf16_t* __restrict__ out,
                              int rows, int cols) {
  int i = blockIdx.x * blockDim.x + threadIdx.x;
  if (i < rows * cols) {
    int r = i % rows;
    int c = i / rows;
    out[i] = (bf16_t)in[(size_t)r * cols + c];
  }
}

// ---- WMMA GEMM: D = A(MxK) * BT(NoutxK)^T ------------------------------
// per wave: 32x64 output tile; 8 waves/block -> 128x128 block tile
__global__ __launch_bounds__(256) void gemm_wmma_bf16out(
    const bf16_t* __restrict__ A, const bf16_t* __restrict__ BT,
    bf16_t* __restrict__ Out, int M, int Nout, int K) {
  const int lane = threadIdx.x & 31;
  const int wave = threadIdx.x >> 5;
  const int m0 = blockIdx.x * 128 + (wave >> 1) * 32;
  const int n0 = blockIdx.y * 128 + (wave & 1) * 64;
  const int mrow = lane & 15;
  const int hx = lane >> 4;

  v8f acc[2][4];
#pragma unroll
  for (int i = 0; i < 2; ++i)
#pragma unroll
    for (int j = 0; j < 4; ++j) acc[i][j] = v8f_zero();

  for (int k0 = 0; k0 < K; k0 += 32) {
    v16bf af[2], bfg[4];
#pragma unroll
    for (int mi = 0; mi < 2; ++mi)
      af[mi] = load_frag(A + (size_t)(m0 + 16 * mi + mrow) * K + k0, hx);
#pragma unroll
    for (int ni = 0; ni < 4; ++ni)
      bfg[ni] = load_frag(BT + (size_t)(n0 + 16 * ni + mrow) * K + k0, hx);
    if (k0 + 32 < K) {
      __builtin_prefetch(A + (size_t)(m0 + mrow) * K + k0 + 32, 0, 1);
      __builtin_prefetch(BT + (size_t)(n0 + mrow) * K + k0 + 32, 0, 1);
    }
#pragma unroll
    for (int mi = 0; mi < 2; ++mi)
#pragma unroll
      for (int ni = 0; ni < 4; ++ni)
        acc[mi][ni] = wmma_bf16(af[mi], bfg[ni], acc[mi][ni]);
  }

#pragma unroll
  for (int mi = 0; mi < 2; ++mi)
#pragma unroll
    for (int ni = 0; ni < 4; ++ni)
#pragma unroll
      for (int r = 0; r < 8; ++r) {
        int row = m0 + 16 * mi + r + 8 * hx;   // C/D layout: row = vgpr + 8*(lane/16)
        int col = n0 + 16 * ni + mrow;          // col = lane%16
        Out[(size_t)row * Nout + col] = (bf16_t)acc[mi][ni][r];
      }
}

__global__ __launch_bounds__(256) void gemm_wmma_f32out_bias(
    const bf16_t* __restrict__ A, const bf16_t* __restrict__ BT,
    const float* __restrict__ bias, float* __restrict__ Out,
    int M, int Nout, int K) {
  const int lane = threadIdx.x & 31;
  const int wave = threadIdx.x >> 5;
  const int m0 = blockIdx.x * 128 + (wave >> 1) * 32;
  const int n0 = blockIdx.y * 128 + (wave & 1) * 64;
  const int mrow = lane & 15;
  const int hx = lane >> 4;

  v8f acc[2][4];
#pragma unroll
  for (int i = 0; i < 2; ++i)
#pragma unroll
    for (int j = 0; j < 4; ++j) acc[i][j] = v8f_zero();

  for (int k0 = 0; k0 < K; k0 += 32) {
    v16bf af[2], bfg[4];
#pragma unroll
    for (int mi = 0; mi < 2; ++mi)
      af[mi] = load_frag(A + (size_t)(m0 + 16 * mi + mrow) * K + k0, hx);
#pragma unroll
    for (int ni = 0; ni < 4; ++ni)
      bfg[ni] = load_frag(BT + (size_t)(n0 + 16 * ni + mrow) * K + k0, hx);
#pragma unroll
    for (int mi = 0; mi < 2; ++mi)
#pragma unroll
      for (int ni = 0; ni < 4; ++ni)
        acc[mi][ni] = wmma_bf16(af[mi], bfg[ni], acc[mi][ni]);
  }

#pragma unroll
  for (int mi = 0; mi < 2; ++mi)
#pragma unroll
    for (int ni = 0; ni < 4; ++ni)
#pragma unroll
      for (int r = 0; r < 8; ++r) {
        int row = m0 + 16 * mi + r + 8 * hx;
        int col = n0 + 16 * ni + mrow;
        Out[(size_t)row * Nout + col] = acc[mi][ni][r] + bias[col];
      }
}

// ---- RoPE + q/k/v relayout --------------------------------------------
// qkv: [b][n][3C] bf16.  q,k -> [b][h][n][64] (rope applied), v -> [b][h][64][n]
__global__ void rope_kernel(const bf16_t* __restrict__ qkv,
                            bf16_t* __restrict__ q, bf16_t* __restrict__ k,
                            bf16_t* __restrict__ vt) {
  int tid = blockIdx.x * blockDim.x + threadIdx.x; // B*N*H*32 threads
  int j = tid & 31;
  int h = (tid >> 5) & (Hh - 1);
  int n = (tid >> 9) & (Nn - 1);
  int b = tid >> 20;
  if (b >= Bz) return;

  size_t base = ((size_t)(b * Nn + n)) * (3 * Cc) + h * Dd;
  float invf = __powf(10000.f, -(float)j * (1.f / 32.f));
  float ang = (float)n * invf;
  float cs = __cosf(ang), sn = __sinf(ang);

  float q1 = (float)qkv[base + 2 * j];
  float q2 = (float)qkv[base + 2 * j + 1];
  float k1 = (float)qkv[base + Cc + 2 * j];
  float k2 = (float)qkv[base + Cc + 2 * j + 1];

  size_t bh = (size_t)b * Hh + h;
  bf16_t* qo = q + (bh * Nn + n) * Dd;
  bf16_t* ko = k + (bh * Nn + n) * Dd;
  qo[j]      = (bf16_t)(q1 * cs - q2 * sn);
  qo[32 + j] = (bf16_t)(q1 * sn + q2 * cs);
  ko[j]      = (bf16_t)(k1 * cs - k2 * sn);
  ko[32 + j] = (bf16_t)(k1 * sn + k2 * cs);

  // v transposed per head: vt[bh][d][n]
  vt[(bh * Dd + j) * Nn + n]      = qkv[base + 2 * Cc + j];
  vt[(bh * Dd + 32 + j) * Nn + n] = qkv[base + 2 * Cc + 32 + j];
}

// ---- causal flash attention (one wave per 16-row q tile) --------------
// q,k: [b][h][n][64] bf16 ; vt: [b][h][64][n] bf16 ; out: [b][n][C] bf16
__global__ __launch_bounds__(128) void attn_kernel(
    const bf16_t* __restrict__ q, const bf16_t* __restrict__ k,
    const bf16_t* __restrict__ vt, bf16_t* __restrict__ attn_out) {
  __shared__ __align__(16) bf16_t plds[4][16 * 32]; // per-wave P staging

  const int lane = threadIdx.x & 31;
  const int w = threadIdx.x >> 5;
  const int tile = blockIdx.x * 4 + w;            // B*H*(N/16) tiles
  const int nt = tile & ((Nn / 16) - 1);
  const int h = (tile >> 7) & (Hh - 1);
  const int b = tile >> 11;
  const int qbase = nt * 16;
  const int mrow = lane & 15;
  const int hx = lane >> 4;

  const size_t bh = (size_t)b * Hh + h;
  const bf16_t* qp = q + (bh * Nn + qbase) * Dd;
  const bf16_t* kp = k + bh * (size_t)Nn * Dd;
  const bf16_t* vp = vt + bh * (size_t)Dd * Nn;

  v16bf qa[2];
#pragma unroll
  for (int s = 0; s < 2; ++s)
    qa[s] = load_frag(qp + (size_t)mrow * Dd + 32 * s, hx);

  float mstat[8], lstat[8];
  v8f acc[4];
#pragma unroll
  for (int r = 0; r < 8; ++r) { mstat[r] = -1e30f; lstat[r] = 0.f; }
#pragma unroll
  for (int dt = 0; dt < 4; ++dt) acc[dt] = v8f_zero();

  for (int kv0 = 0; kv0 < qbase + 16; kv0 += 32) {
    // S = Q * K^T for a 16x32 block of scores (two 16x16 wmma tiles)
    v8f S0 = v8f_zero(), S1 = v8f_zero();
#pragma unroll
    for (int s = 0; s < 2; ++s) {
      v16bf kb0 = load_frag(kp + (size_t)(kv0 + mrow) * Dd + 32 * s, hx);
      v16bf kb1 = load_frag(kp + (size_t)(kv0 + 16 + mrow) * Dd + 32 * s, hx);
      S0 = wmma_bf16(qa[s], kb0, S0);
      S1 = wmma_bf16(qa[s], kb1, S1);
    }

    const int col0 = kv0 + mrow;
    const int col1 = kv0 + 16 + mrow;
#pragma unroll
    for (int r = 0; r < 8; ++r) {
      const int qg = qbase + r + 8 * hx;       // global query row for this slot
      float s0 = S0[r] * SCALE; if (col0 > qg) s0 = -1e30f;
      float s1 = S1[r] * SCALE; if (col1 > qg) s1 = -1e30f;

      float t = fmaxf(s0, s1);
#pragma unroll
      for (int off = 1; off < 16; off <<= 1) t = fmaxf(t, __shfl_xor(t, off, 32));
      float mnew = fmaxf(mstat[r], t);
      float alpha = __expf(mstat[r] - mnew);
      float p0 = __expf(s0 - mnew);
      float p1 = __expf(s1 - mnew);
      float rs = p0 + p1;
#pragma unroll
      for (int off = 1; off < 16; off <<= 1) rs += __shfl_xor(rs, off, 32);
      lstat[r] = lstat[r] * alpha + rs;
      mstat[r] = mnew;
#pragma unroll
      for (int dt = 0; dt < 4; ++dt) acc[dt][r] *= alpha;

      // stage P (C-layout -> memory row-major 16x32) through LDS
      plds[w][(r + 8 * hx) * 32 + mrow] = (bf16_t)p0;
      plds[w][(r + 8 * hx) * 32 + 16 + mrow] = (bf16_t)p1;
    }

    // reload P as an A fragment (row-in-lane) — per-wave LDS, DS ops in-order
    v16bf pa;
    {
      union { v16bf v; v8bf h2[2]; } u;
      u.h2[0] = *(const v8bf*)&plds[w][mrow * 32 + 8 * hx];
      u.h2[1] = *(const v8bf*)&plds[w][mrow * 32 + 16 + 8 * hx];
      pa = u.v;
    }

    // O += P * V  (V columns contiguous thanks to [d][n] layout)
#pragma unroll
    for (int dt = 0; dt < 4; ++dt) {
      v16bf vb = load_frag(vp + (size_t)(dt * 16 + mrow) * Nn + kv0, hx);
      acc[dt] = wmma_bf16(pa, vb, acc[dt]);
    }
  }

#pragma unroll
  for (int dt = 0; dt < 4; ++dt)
#pragma unroll
    for (int r = 0; r < 8; ++r) {
      int tok = qbase + r + 8 * hx;
      float o = acc[dt][r] / lstat[r];
      attn_out[((size_t)b * Nn + tok) * Cc + h * Dd + dt * 16 + mrow] = (bf16_t)o;
    }
}

// ---- launcher ----------------------------------------------------------
extern "C" void kernel_launch(void* const* d_in, const int* in_sizes, int n_in,
                              void* d_out, int out_size, void* d_ws, size_t ws_size,
                              hipStream_t stream) {
  const float* x    = (const float*)d_in[0];
  const float* Wqkv = (const float*)d_in[1];
  const float* Wout = (const float*)d_in[2];
  const float* bout = (const float*)d_in[3];
  float* out = (float*)d_out;

  char* ws = (char*)d_ws;
  size_t off = 0;
  auto carve = [&](size_t bytes) -> void* {
    void* p = ws + off;
    off += (bytes + 255) & ~(size_t)255;
    return p;
  };
  bf16_t* xb   = (bf16_t*)carve((size_t)Mrows * Cc * 2);        // x in bf16
  bf16_t* wqkt = (bf16_t*)carve((size_t)3 * Cc * Cc * 2);       // Wqkv^T bf16
  bf16_t* wot  = (bf16_t*)carve((size_t)Cc * Cc * 2);           // Wout^T bf16
  bf16_t* qkv  = (bf16_t*)carve((size_t)Mrows * 3 * Cc * 2);    // qkv bf16
  bf16_t* qb   = (bf16_t*)carve((size_t)Bz * Hh * Nn * Dd * 2); // q roped
  bf16_t* kb   = (bf16_t*)carve((size_t)Bz * Hh * Nn * Dd * 2); // k roped
  bf16_t* vtb  = (bf16_t*)carve((size_t)Bz * Hh * Dd * Nn * 2); // v transposed
  bf16_t* attn = (bf16_t*)carve((size_t)Mrows * Cc * 2);        // attn out bf16
  (void)ws_size; (void)in_sizes; (void)n_in; (void)out_size;

  // 1) precision conversion + weight transposes
  {
    int n = Mrows * Cc;
    cvt_f32_bf16<<<(n + 255) / 256, 256, 0, stream>>>(x, xb, n);
  }
  transpose_cvt<<<(Cc * 3 * Cc + 255) / 256, 256, 0, stream>>>(Wqkv, wqkt, Cc, 3 * Cc);
  transpose_cvt<<<(Cc * Cc + 255) / 256, 256, 0, stream>>>(Wout, wot, Cc, Cc);

  // 2) qkv = x @ Wqkv    (4096 x 3072, K=1024)
  gemm_wmma_bf16out<<<dim3(Mrows / 128, (3 * Cc) / 128), 256, 0, stream>>>(
      xb, wqkt, qkv, Mrows, 3 * Cc, Cc);

  // 3) RoPE + head-major relayout (+ V transpose)
  {
    int n = Bz * Nn * Hh * 32;
    rope_kernel<<<(n + 255) / 256, 256, 0, stream>>>(qkv, qb, kb, vtb);
  }

  // 4) causal flash attention (one wave per 16-row q tile, 4 waves/block)
  {
    int tiles = Bz * Hh * (Nn / 16); // 4096
    attn_kernel<<<tiles / 4, 128, 0, stream>>>(qb, kb, vtb, attn);
  }

  // 5) out = attn @ Wout + bout  (4096 x 1024, K=1024, f32 out)
  gemm_wmma_f32out_bias<<<dim3(Mrows / 128, Cc / 128), 256, 0, stream>>>(
      attn, wot, bout, out, Mrows, Cc, Cc);
}
